// SNNOnlyModel_65214783422997
// MI455X (gfx1250) — matrix operand order repaired and checked by reading, not compiled, not run on hardware
//
#include <hip/hip_runtime.h>
#include <stdint.h>

typedef _Float16 v16h __attribute__((ext_vector_type(16)));
typedef _Float16 v8h  __attribute__((ext_vector_type(8)));
typedef float    v8f  __attribute__((ext_vector_type(8)));
typedef uint32_t v8u  __attribute__((ext_vector_type(8)));

#define BETA   0.5f
#define THRESH 1.0f
#define NSTEPS 20

static __device__ __forceinline__ v8f wmma_f16(v16h a, v16h b, v8f c) {
    return __builtin_amdgcn_wmma_f32_16x16x32_f16(false, a, false, b,
                                                  (short)0, c, false, false);
}

// ---------------------------------------------------------------------------
// Kernel A: fc1 GEMM via split-precision f16 WMMA (hi+lo covers ~22 mantissa
// bits; aL*bH + aH*bL + aH*bH accumulated in f32) fused with the full 20-step
// LIF-1 dynamics in accumulator registers -> 20-bit spike masks.
//   x [4096,1024] @ W1^T -> cur1 ; spkbits[4096][2048] bit t = spike at step t
// ---------------------------------------------------------------------------
#define KA_BM 128
#define KA_BN 64
#define KA_BK 64
#define AS_STRIDE 72   // halfs per A row (144B: 16B-aligned, banks spread)
#define BS_STRIDE 80   // halfs per B row (160B: 32B-aligned)

__global__ __launch_bounds__(256)
void snn_fc1_kernel(const float* __restrict__ x,
                    const float* __restrict__ W1,
                    const float* __restrict__ b1,
                    uint32_t* __restrict__ spkbits,
                    int M, int N, int K)
{
    __shared__ __attribute__((aligned(32))) _Float16 AsH[KA_BM * AS_STRIDE];
    __shared__ __attribute__((aligned(32))) _Float16 AsL[KA_BM * AS_STRIDE];
    __shared__ __attribute__((aligned(32))) _Float16 BsH[KA_BN * BS_STRIDE];
    __shared__ __attribute__((aligned(32))) _Float16 BsL[KA_BN * BS_STRIDE];

    const int tid  = threadIdx.x;
    const int lane = tid & 31;
    const int wave = tid >> 5;
    const int m0   = blockIdx.y * KA_BM;
    const int n0   = blockIdx.x * KA_BN;
    const int wm   = wave * 16;
    const int mf   = lane & 15;        // A row / B col / C col lane component
    const int hh   = lane >> 4;        // lane half selector

    v8f acc[4] = {};                   // 16 x 64 f32 output per wave

    for (int kk = 0; kk < K; kk += KA_BK) {
        // stage A chunk: 128 x 64 f32 -> hi/lo halfs; 32 contiguous per thread
        {
            int mrow = tid >> 1;
            int kcol = (tid & 1) * 32;
            const float* src = x + (size_t)(m0 + mrow) * K + kk + kcol;
            _Float16* dh = AsH + mrow * AS_STRIDE + kcol;
            _Float16* dl = AsL + mrow * AS_STRIDE + kcol;
            #pragma unroll
            for (int j = 0; j < 32; ++j) {
                float w = src[j];
                _Float16 hi = (_Float16)w;
                dh[j] = hi;
                dl[j] = (_Float16)(w - (float)hi);
            }
        }
        // stage B chunk: B[n][k] = W1[n0+n][kk+k]; 64 x 64, 16 per thread
        {
            int nrow = tid >> 2;
            int kcol = (tid & 3) * 16;
            const float* src = W1 + (size_t)(n0 + nrow) * K + kk + kcol;
            _Float16* dh = BsH + nrow * BS_STRIDE + kcol;
            _Float16* dl = BsL + nrow * BS_STRIDE + kcol;
            #pragma unroll
            for (int j = 0; j < 16; ++j) {
                float w = src[j];
                _Float16 hi = (_Float16)w;
                dh[j] = hi;
                dl[j] = (_Float16)(w - (float)hi);
            }
        }
        __syncthreads();

        #pragma unroll
        for (int g = 0; g < 2; ++g) {          // two K=32 WMMA groups
            // A-fragment: K = 32g + 16*hh + j, 16 contiguous halfs
            const _Float16* arh = AsH + (wm + mf) * AS_STRIDE + 32 * g;
            const _Float16* arl = AsL + (wm + mf) * AS_STRIDE + 32 * g;
            v8h h0 = *(const v8h*)(arh + 16 * hh);
            v8h h1 = *(const v8h*)(arh + 16 * hh + 8);
            v8h l0 = *(const v8h*)(arl + 16 * hh);
            v8h l1 = *(const v8h*)(arl + 16 * hh + 8);
            v16h aH = __builtin_shufflevector(h0, h1, 0,1,2,3,4,5,6,7,
                                              8,9,10,11,12,13,14,15);
            v16h aL = __builtin_shufflevector(l0, l1, 0,1,2,3,4,5,6,7,
                                              8,9,10,11,12,13,14,15);
            #pragma unroll
            for (int ns = 0; ns < 4; ++ns) {
                // B-fragment: K = 32g + 16*hh + j, 16 contiguous halfs
                size_t boff = (size_t)(ns * 16 + mf) * BS_STRIDE + 32 * g + 16 * hh;
                v16h bH = *(const v16h*)(BsH + boff);
                v16h bL = *(const v16h*)(BsL + boff);
                acc[ns] = wmma_f16(aL, bH, acc[ns]);
                acc[ns] = wmma_f16(aH, bL, acc[ns]);
                acc[ns] = wmma_f16(aH, bH, acc[ns]);
            }
        }
        __syncthreads();
    }

    // Epilogue: bias + 20-step LIF entirely in registers -> spike bitmask
    #pragma unroll
    for (int ns = 0; ns < 4; ++ns) {
        int n = n0 + ns * 16 + mf;
        float bias = b1[n];
        #pragma unroll
        for (int v = 0; v < 8; ++v) {
            int m = m0 + wm + v + hh * 8;      // C/D layout row
            float cur = acc[ns][v] + bias;
            float u = 0.0f;
            uint32_t bits = 0u;
            #pragma unroll
            for (int t = 0; t < NSTEPS; ++t) {
                u = BETA * u + cur;
                if (u > THRESH) { bits |= (1u << t); u -= THRESH; }
            }
            spkbits[(size_t)m * N + n] = bits;
        }
    }
}

// ---------------------------------------------------------------------------
// Pack W2 [C][H] into split-precision f16 in B-fragment-native order:
//   pack[(k/32)*(C*32) + c*32 + (k%32)]  (lane reads 16 contiguous halfs)
// ---------------------------------------------------------------------------
__global__ __launch_bounds__(256)
void pack_w2_kernel(const float* __restrict__ W2,
                    _Float16* __restrict__ hi, _Float16* __restrict__ lo,
                    int C, int H)
{
    int idx = blockIdx.x * 256 + threadIdx.x;
    if (idx >= C * H) return;
    int c = idx / H;
    int k = idx - c * H;
    float w = W2[idx];
    _Float16 h = (_Float16)w;
    size_t p = (size_t)(k >> 5) * ((size_t)C * 32) + (size_t)c * 32 + (k & 31);
    hi[p] = h;
    lo[p] = (_Float16)(w - (float)h);
}

// ---------------------------------------------------------------------------
// Kernel B: per 16-batch-row tile, 20 steps of
//   cur2(t) = spk1(t) @ W2^T + b2  via f16 WMMA (A from spike bits, exact
//   {0,1}; B = W2lo then W2hi, both accumulated into the same f32 C)
//   mem2 LIF + spike accumulation live in WMMA C-layout registers.
// 8 waves x 2 N-tiles cover C=256. No LDS; waves independent.
// ---------------------------------------------------------------------------
__global__ __launch_bounds__(256)
void snn_fc2_kernel(const uint32_t* __restrict__ spkbits,
                    const _Float16* __restrict__ W2hi,
                    const _Float16* __restrict__ W2lo,
                    const float* __restrict__ b2,
                    float* __restrict__ out,
                    int H, int C)
{
    const int tid  = threadIdx.x;
    const int lane = tid & 31;
    const int wave = tid >> 5;
    const int b0   = blockIdx.x * 16;
    const int mf   = lane & 15;
    const int hh   = lane >> 4;
    const int c0   = wave * 32;

    float b2v[2];
    b2v[0] = b2[c0 + mf];
    b2v[1] = b2[c0 + 16 + mf];

    v8f mem2[2] = {};
    v8f accs[2] = {};

    const uint32_t* arow = spkbits + (size_t)(b0 + mf) * H;

    for (int t = 0; t < NSTEPS; ++t) {
        v8f acc2[2] = {};
        for (int k = 0; k < H; k += 32) {
            // A-fragment spike words: K = 16*hh + j
            uint4 q0 = *(const uint4*)(arow + k + 16 * hh);
            uint4 q1 = *(const uint4*)(arow + k + 16 * hh + 4);
            uint4 q2 = *(const uint4*)(arow + k + 16 * hh + 8);
            uint4 q3 = *(const uint4*)(arow + k + 16 * hh + 12);
            uint32_t w[16] = { q0.x, q0.y, q0.z, q0.w,  q1.x, q1.y, q1.z, q1.w,
                               q2.x, q2.y, q2.z, q2.w,  q3.x, q3.y, q3.z, q3.w };
            v8u ap;
            #pragma unroll
            for (int i = 0; i < 8; ++i) {
                uint32_t e = (w[2 * i]     >> t) & 1u;
                uint32_t o = (w[2 * i + 1] >> t) & 1u;
                ap[i] = e * 0x3C00u + o * 0x3C000000u;   // pair of f16 {0,1}
            }
            v16h a = __builtin_bit_cast(v16h, ap);

            size_t chunk_base = (size_t)(k >> 5) * ((size_t)C * 32);
            if (k + 32 < H)
                __builtin_prefetch(W2hi + chunk_base + (size_t)C * 32
                                        + (size_t)(c0 + mf) * 32, 0, 3);
            #pragma unroll
            for (int ci = 0; ci < 2; ++ci) {
                size_t boff = chunk_base + (size_t)(c0 + ci * 16 + mf) * 32 + 16 * hh;
                v16h bH = *(const v16h*)(W2hi + boff);
                v16h bL = *(const v16h*)(W2lo + boff);
                acc2[ci] = wmma_f16(a, bL, acc2[ci]);
                acc2[ci] = wmma_f16(a, bH, acc2[ci]);
            }
        }
        #pragma unroll
        for (int ci = 0; ci < 2; ++ci) {
            #pragma unroll
            for (int v = 0; v < 8; ++v) {
                float m2 = BETA * mem2[ci][v] + acc2[ci][v] + b2v[ci];
                if (m2 > THRESH) { m2 -= THRESH; accs[ci][v] += 1.0f; }
                mem2[ci][v] = m2;
            }
        }
    }

    #pragma unroll
    for (int ci = 0; ci < 2; ++ci) {
        int c = c0 + ci * 16 + mf;
        #pragma unroll
        for (int v = 0; v < 8; ++v) {
            int m = b0 + v + hh * 8;
            out[(size_t)m * C + c] = accs[ci][v] * (1.0f / NSTEPS);
        }
    }
}

// ---------------------------------------------------------------------------
extern "C" void kernel_launch(void* const* d_in, const int* in_sizes, int n_in,
                              void* d_out, int out_size, void* d_ws, size_t ws_size,
                              hipStream_t stream) {
    const float* x  = (const float*)d_in[0];   // [4096, 64, 16] -> [4096,1024]
    const float* W1 = (const float*)d_in[1];   // [2048, 1024]
    const float* b1 = (const float*)d_in[2];   // [2048]
    const float* W2 = (const float*)d_in[3];   // [256, 2048]
    const float* b2 = (const float*)d_in[4];   // [256]
    float* out = (float*)d_out;                // [4096, 256]

    const int B = 4096, IN = 1024, H = 2048, C = 256;

    // workspace: spike masks [B][H] u32 ; W2 hi/lo packs [H/32][C][32] f16
    uint32_t* spkbits = (uint32_t*)d_ws;
    char* p1 = (char*)d_ws + (size_t)B * H * sizeof(uint32_t);
    _Float16* W2hi = (_Float16*)p1;
    _Float16* W2lo = (_Float16*)(p1 + (size_t)H * C * sizeof(_Float16));

    dim3 gA(H / KA_BN, B / KA_BM);   // (32, 32)
    snn_fc1_kernel<<<gA, 256, 0, stream>>>(x, W1, b1, spkbits, B, H, IN);

    pack_w2_kernel<<<(C * H + 255) / 256, 256, 0, stream>>>(W2, W2hi, W2lo, C, H);

    snn_fc2_kernel<<<B / 16, 256, 0, stream>>>(spkbits, W2hi, W2lo, b2, out, H, C);
}
